// single_head_local_attention_4818953307054
// MI455X (gfx1250) — matrix operand order repaired and checked by reading, not compiled, not run on hardware
//
#include <hip/hip_runtime.h>

// Single-head local (windowed) attention, kernel=5, pad=2, T=16.
// Shapes hardcoded from the reference: B=8, C=256, H=W=96.

constexpr int Bn = 8, Cn = 256, Hn = 96, Wn = 96;
constexpr int KW = 5, PAD = 2;
constexpr int TX = 16;     // spatial x-tile per block
constexpr int NT = 160;    // 5 wave32 waves: wave w handles dy = w
constexpr float TEMP_INV = 1.0f / 16.0f;

typedef __attribute__((ext_vector_type(16))) __bf16 v16bf;
typedef __attribute__((ext_vector_type(8)))  float  v8f;

// float -> bf16 round-to-nearest-even in integer math.
__device__ __forceinline__ unsigned short f2bf(float f) {
    unsigned int u = __float_as_uint(f);
    u += 0x7FFFu + ((u >> 16) & 1u);
    return (unsigned short)(u >> 16);
}

// Low 32 bits of a generic pointer to LDS == LDS byte offset (ISA: shared
// aperture keeps the LDS offset in addr[31:0]).
__device__ __forceinline__ unsigned int lds_off(const void* p) {
    return (unsigned int)(unsigned long long)(uintptr_t)p;
}

// CDNA5 async global->LDS copies (tracked by ASYNCcnt, per-lane, EXEC-masked).
__device__ __forceinline__ void async_g2l_b128(unsigned int lds_addr, const void* g) {
    asm volatile("global_load_async_to_lds_b128 %0, %1, off"
                 :: "v"(lds_addr), "v"((unsigned long long)(uintptr_t)g)
                 : "memory");
}
__device__ __forceinline__ void async_g2l_b64(unsigned int lds_addr, const void* g) {
    asm volatile("global_load_async_to_lds_b64 %0, %1, off"
                 :: "v"(lds_addr), "v"((unsigned long long)(uintptr_t)g)
                 : "memory");
}
__device__ __forceinline__ void wait_async0() {
    asm volatile("s_wait_asynccnt 0x0" ::: "memory");
}

// A-matrix (16x32 bf16) per-lane fragment: lanes 0-15 hold K=0..15
// contiguous, lanes 16-31 hold K=16..31.  p = &row[s*32 + hi*16].
__device__ __forceinline__ v16bf lds_load_a(const unsigned short* p) {
    union { v16bf v; uint4 u[2]; } r;
    r.u[0] = *(const uint4*)(p);
    r.u[1] = *(const uint4*)(p + 8);
    return r.v;
}
// B-matrix (32x16 bf16) per-lane fragment: lanes 0-15 hold K=0..15
// contiguous, lanes 16-31 hold K=16..31.  p = &col[s*32 + hi*16].
__device__ __forceinline__ v16bf lds_load_b(const unsigned short* p) {
    union { v16bf v; uint4 u[2]; } r;
    r.u[0] = *(const uint4*)(p);
    r.u[1] = *(const uint4*)(p + 8);
    return r.v;
}

__global__ __launch_bounds__(NT) void slha_wmma_kernel(
    const float* __restrict__ q, const float* __restrict__ k,
    const float* __restrict__ v, float* __restrict__ out)
{
    // fp32 async staging buffers
    __shared__ __align__(16) float qstage[64][TX];        //  4 KB  [c][x]
    __shared__ __align__(16) float kstage[KW][64][20];    // 25.6KB [dy][c][x']
    // bf16 WMMA operand tiles (ushort-typed to avoid scalar __bf16 codegen)
    __shared__ __align__(16) unsigned short q_s[TX][64];      // [x][c]
    __shared__ __align__(16) unsigned short k_s[KW][32][64];  // [dy][x'][c]
    __shared__ float w_s[TX][32];

    const int tx   = blockIdx.x;          // 0..5
    const int y    = blockIdx.y;          // 0..95
    const int b    = blockIdx.z;          // 0..7
    const int x0   = tx * TX;
    const int t    = threadIdx.x;
    const int lane = t & 31;
    const int wave = t >> 5;              // == dy (0..4)
    const int lr   = lane & 15;
    const int hi   = lane >> 4;

    // Whole 20-wide k window in x-range? (x0-2 >= 0 && x0+17 < Wn)
    const bool interior = (x0 >= PAD) && (x0 + TX - 1 + PAD < Wn);

    // Columns 20..31 of k_s feed only discarded band entries; zero them once.
    for (int i = t; i < KW * 12 * 64; i += NT) {
        int xl = 20 + (i % 12);
        int c  = (i / 12) % 64;
        int dy = i / (12 * 64);
        k_s[dy][xl][c] = 0;
    }

    v8f acc0 = {};   // band tile: x' cols 0..15  (global x0-2 .. x0+13)
    v8f acc1 = {};   // band tile: x' cols 16..31 (global x0+14 .. x0+29)

    const unsigned int qs_base = lds_off(&qstage[0][0]);
    const unsigned int ks_base = lds_off(&kstage[0][0][0]);

    // ---- Phase 1: corr = q . shifted(k) over C=256 via bf16 WMMA ----
    for (int ch = 0; ch < Cn / 64; ++ch) {
        const int cb = ch * 64;

        // q tile: async b128, 4 copies per channel (always fully in range)
        for (int i = t; i < 64 * 4; i += NT) {
            int xq = i & 3, c = i >> 2;
            async_g2l_b128(qs_base + (unsigned)(c * TX + xq * 4) * 4u,
                           &q[((b * Cn + cb + c) * Hn + y) * Wn + x0 + xq * 4]);
        }
        if (interior) {
            // k window: async b64, 10 copies per (dy, channel) row
            for (int i = t; i < KW * 64 * 10; i += NT) {
                int xp = i % 10;
                int c  = (i / 10) % 64;
                int dy = i / 640;
                int yy = y + dy - PAD;
                if (yy >= 0 && yy < Hn)
                    async_g2l_b64(ks_base + (unsigned)((dy * 64 + c) * 20 + xp * 2) * 4u,
                                  &k[((b * Cn + cb + c) * Hn + yy) * Wn + (x0 - PAD) + xp * 2]);
            }
        } else {
            // edge x-tiles: masked scalar staging (OOB entries masked later)
            for (int i = t; i < KW * 64 * 20; i += NT) {
                int xl = i % 20;
                int c  = (i / 20) % 64;
                int dy = i / (20 * 64);
                int yy = y + dy - PAD;
                int xx = x0 - PAD + xl;
                if (yy >= 0 && yy < Hn && xx >= 0 && xx < Wn)
                    kstage[dy][c][xl] = k[((b * Cn + cb + c) * Hn + yy) * Wn + xx];
            }
        }
        wait_async0();
        __syncthreads();

        // Convert + transpose into bf16 WMMA operand layouts
        for (int i = t; i < TX * 64; i += NT) {
            int x = i & 15, c = i >> 4;
            q_s[x][c] = f2bf(qstage[c][x]);
        }
        for (int i = t; i < KW * 64 * 20; i += NT) {
            int xl = i % 20;
            int c  = (i / 20) % 64;
            int dy = i / (20 * 64);
            int yy = y + dy - PAD;
            int xx = x0 - PAD + xl;
            float val = 0.0f;
            if (yy >= 0 && yy < Hn && xx >= 0 && xx < Wn) val = kstage[dy][c][xl];
            k_s[dy][xl][c] = f2bf(val);
        }
        __syncthreads();

        #pragma unroll
        for (int s = 0; s < 2; ++s) {
            v16bf a  = lds_load_a(&q_s[lr][s * 32 + hi * 16]);
            v16bf b0 = lds_load_b(&k_s[wave][lr][s * 32 + hi * 16]);
            v16bf b1 = lds_load_b(&k_s[wave][16 + lr][s * 32 + hi * 16]);
            acc0 = __builtin_amdgcn_wmma_f32_16x16x32_bf16(
                false, a, false, b0, (short)0, acc0, false, false);
            acc1 = __builtin_amdgcn_wmma_f32_16x16x32_bf16(
                false, a, false, b1, (short)0, acc1, false, false);
        }
        __syncthreads();
    }

    // ---- Extract the 5-wide diagonal band, apply ReLU ----
    // C/D layout: VGPR r, lane: M = r + 8*hi (x-local), N = lr.
    // Reference stores corr at idx = dx*5 + dy.
    #pragma unroll
    for (int r = 0; r < 8; ++r) {
        int M   = r + 8 * hi;
        int dx0 = lr - M;
        if (dx0 >= 0 && dx0 < KW)
            w_s[M][dx0 * KW + wave] = fmaxf(acc0[r], 0.0f);
        int dx1 = lr + 16 - M;
        if (dx1 >= 0 && dx1 < KW)
            w_s[M][dx1 * KW + wave] = fmaxf(acc1[r], 0.0f);
    }
    __syncthreads();

    // ---- Softmax over the 25 window scores (per pixel) ----
    if (t < TX) {
        float m = 0.0f;                       // ReLU guarantees scores >= 0
        #pragma unroll
        for (int j = 0; j < KW * KW; ++j) m = fmaxf(m, w_s[t][j]);
        float e[KW * KW];
        float ssum = 0.0f;
        #pragma unroll
        for (int j = 0; j < KW * KW; ++j) {
            e[j] = __expf((w_s[t][j] - m) * TEMP_INV);
            ssum += e[j];
        }
        float inv = 1.0f / ssum;
        #pragma unroll
        for (int j = 0; j < KW * KW; ++j) w_s[t][j] = e[j] * inv;
    }
    __syncthreads();

    // ---- Apply: out[b,c,y,x] = sum_{ky,kx} w[ky*5+kx] * vp(y+ky-2, x+kx-2)
    // NT=160 => i&15 is thread-invariant: hoist weights and x-bounds.
    {
        const int x = t & 15;
        float wreg[KW * KW];
        #pragma unroll
        for (int j = 0; j < KW * KW; ++j) wreg[j] = w_s[x][j];

        int  xxv[KW];
        bool xok[KW];
        #pragma unroll
        for (int kx = 0; kx < KW; ++kx) {
            xxv[kx] = x0 + x + kx - PAD;
            xok[kx] = (xxv[kx] >= 0) && (xxv[kx] < Wn);
        }

        for (int c = t >> 4; c < Cn; c += NT / TX) {
            const float* vbase = &v[((b * Cn + c) * Hn) * Wn];
            float accv = 0.0f;
            #pragma unroll
            for (int ky = 0; ky < KW; ++ky) {
                int yy = y + ky - PAD;
                if (yy < 0 || yy >= Hn) continue;
                const float* vrow = vbase + yy * Wn;
                #pragma unroll
                for (int kx = 0; kx < KW; ++kx) {
                    if (xok[kx]) accv += wreg[ky * KW + kx] * vrow[xxv[kx]];
                }
            }
            out[((b * Cn + c) * Hn + y) * Wn + x0 + x] = accv;
        }
    }
}

extern "C" void kernel_launch(void* const* d_in, const int* in_sizes, int n_in,
                              void* d_out, int out_size, void* d_ws, size_t ws_size,
                              hipStream_t stream) {
    (void)in_sizes; (void)n_in; (void)out_size; (void)d_ws; (void)ws_size;
    const float* q = (const float*)d_in[0];
    const float* k = (const float*)d_in[1];
    const float* v = (const float*)d_in[2];
    float* out = (float*)d_out;

    dim3 grid(Wn / TX, Hn, Bn);   // (6, 96, 8)
    slha_wmma_kernel<<<grid, NT, 0, stream>>>(q, k, v, out);
}
